// Decoder_12979391169156
// MI455X (gfx1250) — matrix operand order repaired and checked by reading, not compiled, not run on hardware
//
#include <hip/hip_runtime.h>
#include <hip/hip_bf16.h>

typedef __attribute__((ext_vector_type(16))) _Float16 v16h;
typedef __attribute__((ext_vector_type(8)))  float    v8f;

#define DIMK   256
#define NROWS  1024            // B*S = 16*64
#define VOC    50000
#define COLS_PER_WG 128        // 8 waves * 16 cols
#define NCHUNKS ((VOC + COLS_PER_WG - 1) / COLS_PER_WG)   // 391
#define SF_     48
#define B_      16
#define S_      64

__device__ __forceinline__ float neg_inf() { return -__builtin_inff(); }

// ---------------------------------------------------------------- z + KL ----
__global__ __launch_bounds__(256)
void zkl_kernel(const float* __restrict__ mu, const float* __restrict__ sg,
                float* __restrict__ z, float* __restrict__ kl_part, int n) {
    int i = blockIdx.x * blockDim.x + threadIdx.x;
    float v = 0.0f;
    if (i < n) {
        float m = mu[i], s = sg[i];
        z[i] = m + s;
        v = -logf(s) + 0.5f * (s * s + m * m) - 0.5f;
    }
    __shared__ float red[256];
    red[threadIdx.x] = v;
    __syncthreads();
    for (int off = 128; off > 0; off >>= 1) {
        if (threadIdx.x < off) red[threadIdx.x] += red[threadIdx.x + off];
        __syncthreads();
    }
    if (threadIdx.x == 0) kl_part[blockIdx.x] = red[0];
}

// ------------------------------------------ WMMA GEMM + softmax partials ----
// grid: NCHUNKS blocks of 256 threads (8 waves). Each wave owns 16 vocab
// columns; its B-operand (16 W rows x K=256 in f16) is loaded and converted
// ONCE into 64 VGPRs, then all 64 row-blocks of z are streamed through it.
// partials[row][chunk][2] = (rowmax, sumexp)
#define CVT4(dst, off, f4)                                     \
    dst[(off) + 0] = (_Float16)(f4).x;                         \
    dst[(off) + 1] = (_Float16)(f4).y;                         \
    dst[(off) + 2] = (_Float16)(f4).z;                         \
    dst[(off) + 3] = (_Float16)(f4).w;

__global__ __launch_bounds__(256)
void gemm_softmax_partial(const float* __restrict__ z,
                          const float* __restrict__ W,
                          const float* __restrict__ bias,
                          int nvocab, float* __restrict__ partials) {
    __shared__ float lds_max[8][16];
    __shared__ float lds_sum[8][16];

    const int chunk = blockIdx.x;
    const int w     = threadIdx.x >> 5;
    const int lane  = threadIdx.x & 31;
    const int lL    = lane & 15;
    const int hi    = lane >> 4;

    const int  col      = chunk * COLS_PER_WG + w * 16 + lL;
    const bool colvalid = (col < nvocab);
    const int  wrow     = colvalid ? col : (nvocab - 1);
    const float bval    = colvalid ? bias[col] : 0.0f;

    // ---- load + convert this wave's whole B operand once (64 VGPRs) ----
    // B fragment: lane lo -> K {0..7,16..23}, hi -> K {8..15,24..31} of W row (per 32-K step)
    const float* wbase = W + (size_t)wrow * DIMK + hi * 8;
    v16h bf[8];
#pragma unroll
    for (int kb = 0; kb < 8; ++kb) {
        const float4* wp = reinterpret_cast<const float4*>(wbase + kb * 32);
        float4 b0 = wp[0], b1 = wp[1], b2 = wp[4], b3 = wp[5];
        CVT4(bf[kb], 0, b0)  CVT4(bf[kb], 4, b1)
        CVT4(bf[kb], 8, b2)  CVT4(bf[kb], 12, b3)
    }

    // ---- stream all 1024 rows of z through the register-resident B ----
    for (int mb = 0; mb < 64; ++mb) {
        const int m0 = mb * 16;
        // A fragment: lane lo -> K {0..7,16..23}, hi -> K {8..15,24..31}
        const float* zbase = z + (size_t)(m0 + lL) * DIMK + hi * 8;
        if (mb < 63)  // prefetch next row-block's z tile (global_prefetch_b8)
            __builtin_prefetch(zbase + 16 * DIMK);

        v8f acc = {};
#pragma unroll
        for (int kb = 0; kb < 8; ++kb) {
            const float4* zp = reinterpret_cast<const float4*>(zbase + kb * 32);
            float4 a0 = zp[0], a1 = zp[1];   // k0 + hi*8 + 0..7
            float4 a2 = zp[4], a3 = zp[5];   // k0 + 16 + hi*8 + 0..7
            v16h a;
            CVT4(a, 0, a0)  CVT4(a, 4, a1)  CVT4(a, 8, a2)  CVT4(a, 12, a3)
            acc = __builtin_amdgcn_wmma_f32_16x16x32_f16(
                false, a, false, bf[kb], (short)0, acc, false, false);
        }

        // acc[r]: lanes 0-15 = row m0+r, lanes 16-31 = row m0+r+8; col = lane%16
        float rmaxv[8], rsumv[8];
#pragma unroll
        for (int r = 0; r < 8; ++r) {
            float v = colvalid ? (acc[r] + bval) : neg_inf();
            float m = v;
#pragma unroll
            for (int sft = 1; sft < 16; sft <<= 1)
                m = fmaxf(m, __shfl_xor(m, sft, 32));
            float e = (colvalid && m > neg_inf()) ? expf(v - m) : 0.0f;
#pragma unroll
            for (int sft = 1; sft < 16; sft <<= 1)
                e += __shfl_xor(e, sft, 32);
            rmaxv[r] = m; rsumv[r] = e;
        }
        if (lane == 0 || lane == 16) {
#pragma unroll
            for (int r = 0; r < 8; ++r) {
                lds_max[w][r + hi * 8] = rmaxv[r];
                lds_sum[w][r + hi * 8] = rsumv[r];
            }
        }
        __syncthreads();
        if (threadIdx.x < 16) {
            float M = neg_inf();
            for (int ww = 0; ww < 8; ++ww) M = fmaxf(M, lds_max[ww][threadIdx.x]);
            float Ssum = 0.0f;
            if (M > neg_inf())
                for (int ww = 0; ww < 8; ++ww)
                    Ssum += lds_sum[ww][threadIdx.x] * expf(lds_max[ww][threadIdx.x] - M);
            size_t row = (size_t)m0 + threadIdx.x;
            partials[(row * NCHUNKS + chunk) * 2 + 0] = M;
            partials[(row * NCHUNKS + chunk) * 2 + 1] = Ssum;
        }
        __syncthreads();
    }
}

// ------------------------------------------------- logsumexp finalize -------
__global__ __launch_bounds__(32)
void lse_kernel(const float* __restrict__ partials, float* __restrict__ lse) {
    int row = blockIdx.x, lane = threadIdx.x;
    float M = neg_inf(), Ssum = 0.0f;
    for (int c = lane; c < NCHUNKS; c += 32) {
        float pm = partials[((size_t)row * NCHUNKS + c) * 2 + 0];
        float ps = partials[((size_t)row * NCHUNKS + c) * 2 + 1];
        if (pm > M) { Ssum = Ssum * expf(M - pm) + ps; M = pm; }
        else if (pm > neg_inf()) Ssum += ps * expf(pm - M);
    }
    for (int sft = 1; sft < 32; sft <<= 1) {
        float oM = __shfl_xor(M, sft, 32);
        float oS = __shfl_xor(Ssum, sft, 32);
        if (oM > M) { Ssum = Ssum * expf(M - oM) + oS; M = oM; }
        else if (oM > neg_inf()) Ssum += oS * expf(oM - M);
    }
    if (lane == 0) lse[row] = M + logf(Ssum);
}

// ------------------------------------------------ gathered target logits ----
__global__ __launch_bounds__(256)
void sel_e_kernel(const float* __restrict__ z, const float* __restrict__ W,
                  const float* __restrict__ bias, const int* __restrict__ eng,
                  float* __restrict__ out) {
    int gw = (blockIdx.x * blockDim.x + threadIdx.x) >> 5;   // 1024 waves exact
    int lane = threadIdx.x & 31;
    int idx = eng[gw];
    const float* zr = z + (size_t)gw * DIMK;
    const float* wr = W + (size_t)idx * DIMK;
    float d = 0.0f;
    for (int k = lane; k < DIMK; k += 32) d += zr[k] * wr[k];
    for (int sft = 1; sft < 32; sft <<= 1) d += __shfl_xor(d, sft, 32);
    if (lane == 0) out[gw] = d + bias[idx];
}

__global__ __launch_bounds__(256)
void sel_f_kernel(const float* __restrict__ z, const float* __restrict__ W,
                  const float* __restrict__ bias, const int* __restrict__ fre,
                  float* __restrict__ out) {
    int gw = (blockIdx.x * blockDim.x + threadIdx.x) >> 5;   // 49152 waves exact
    int lane = threadIdx.x & 31;
    int row = gw / SF_, j = gw % SF_;
    int b = row / S_;
    int idx = fre[b * SF_ + j];
    const float* zr = z + (size_t)row * DIMK;
    const float* wr = W + (size_t)idx * DIMK;
    float d = 0.0f;
    for (int k = lane; k < DIMK; k += 32) d += zr[k] * wr[k];
    for (int sft = 1; sft < 32; sft <<= 1) d += __shfl_xor(d, sft, 32);
    if (lane == 0) out[gw] = d + bias[idx];
}

// ---------------------------------------------------------------- final -----
__global__ __launch_bounds__(256)
void final_kernel(const float* __restrict__ sel_e, const float* __restrict__ lse_e,
                  const float* __restrict__ sel_f, const float* __restrict__ lse_f,
                  const float* __restrict__ kl_part, float* __restrict__ out) {
    __shared__ float red[256];
    float lk = 0.0f;
    for (int r = threadIdx.x; r < NROWS; r += 256) lk += sel_e[r] - lse_e[r];
    for (int t = threadIdx.x; t < B_ * SF_; t += 256) {
        int b = t / SF_, j = t % SF_;
        float acc = 0.0f;
        for (int s = 0; s < S_; ++s) {
            int row = b * S_ + s;
            acc += expf(sel_f[row * SF_ + j] - lse_f[row]);
        }
        lk += logf(acc * (1.0f / (float)S_));
    }
    red[threadIdx.x] = lk;
    __syncthreads();
    for (int off = 128; off > 0; off >>= 1) {
        if (threadIdx.x < off) red[threadIdx.x] += red[threadIdx.x + off];
        __syncthreads();
    }
    if (threadIdx.x == 0) out[0] = red[0];
    __syncthreads();
    float kl = 0.0f;
    for (int r = threadIdx.x; r < 1024; r += 256) kl += kl_part[r];
    red[threadIdx.x] = kl;
    __syncthreads();
    for (int off = 128; off > 0; off >>= 1) {
        if (threadIdx.x < off) red[threadIdx.x] += red[threadIdx.x + off];
        __syncthreads();
    }
    if (threadIdx.x == 0) out[1] = red[0];
}

// ---------------------------------------------------------------- launch ----
extern "C" void kernel_launch(void* const* d_in, const int* in_sizes, int n_in,
                              void* d_out, int out_size, void* d_ws, size_t ws_size,
                              hipStream_t stream) {
    const float* mu  = (const float*)d_in[0];
    const float* sg  = (const float*)d_in[1];
    const int*   eng = (const int*)  d_in[2];
    const int*   fre = (const int*)  d_in[3];
    const float* We  = (const float*)d_in[4];
    const float* be  = (const float*)d_in[5];
    const float* Wf  = (const float*)d_in[6];
    const float* bf  = (const float*)d_in[7];
    float* out = (float*)d_out;
    float* ws  = (float*)d_ws;

    const int NZ = NROWS * DIMK;                 // 262144
    // workspace layout (floats)
    float* z       = ws;                         // 262144
    float* kl_part = z + NZ;                     // 1024
    float* lse_e   = kl_part + 1024;             // 1024
    float* lse_f   = lse_e + 1024;               // 1024
    float* sel_e   = lse_f + 1024;               // 1024
    float* sel_f   = sel_e + 1024;               // 49152
    float* part_e  = sel_f + NROWS * SF_;        // 1024*391*2
    float* part_f  = part_e + (size_t)NROWS * NCHUNKS * 2;

    zkl_kernel<<<NZ / 256, 256, 0, stream>>>(mu, sg, z, kl_part, NZ);

    gemm_softmax_partial<<<NCHUNKS, 256, 0, stream>>>(z, We, be, VOC, part_e);
    gemm_softmax_partial<<<NCHUNKS, 256, 0, stream>>>(z, Wf, bf, VOC, part_f);

    lse_kernel<<<NROWS, 32, 0, stream>>>(part_e, lse_e);
    lse_kernel<<<NROWS, 32, 0, stream>>>(part_f, lse_f);

    sel_e_kernel<<<NROWS / 8, 256, 0, stream>>>(z, We, be, eng, sel_e);
    sel_f_kernel<<<NROWS * SF_ / 8, 256, 0, stream>>>(z, Wf, bf, fre, sel_f);

    final_kernel<<<1, 256, 0, stream>>>(sel_e, lse_e, sel_f, lse_f, kl_part, out);
}
